// MultiDepthwiseXcorr_4440996184561
// MI455X (gfx1250) — compile-verified, matrix operand order checked by reading, not run on hardware
//
#include <hip/hip_runtime.h>
#include <hip/hip_bf16.h>
#include <stdint.h>

// ---------------------------------------------------------------------------
// MI455X (gfx1250): bf16 WMMA implicit-GEMM convs, fused BN/ReLU epilogues,
// VALU depthwise multi-xcorr. wave32, v_wmma_f32_16x16x32_bf16.
// Each wave computes a 16x64 output strip: 4 independent accumulators
// (breaks the WMMA->WMMA RAW hazard chain), shared A fragment with
// matrix_a_reuse, and a ping/pong double-buffered K-loop (unrolled x2, no
// register copies) so global-load latency hides behind the WMMA groups.
// ---------------------------------------------------------------------------

typedef __attribute__((ext_vector_type(16))) __bf16 v16bf;
typedef __attribute__((ext_vector_type(8)))  __bf16 v8bf;
typedef __attribute__((ext_vector_type(8)))  float  v8f;

#define BSZ   64
#define CIN   256
#define HID   256
#define COUTN 10
#define EPSV  1e-5f

// 16-element bf16 fragment from two contiguous 8-element chunks.
// ISA 16-bit 16x32 layout: lane L holds row/col (L%16); K values
// k0 + 8*(L/16) + j (VGPRs 0-3) and k0 + 16 + 8*(L/16) + j (VGPRs 4-7).
__device__ __forceinline__ v16bf frag_cat(const __bf16* p) {
  v8bf c0 = *(const v8bf*)p;
  v8bf c1 = *(const v8bf*)(p + 16);
  v16bf r;
#pragma unroll
  for (int j = 0; j < 8; ++j) { r[j] = c0[j]; r[8 + j] = c1[j]; }
  return r;
}

#define WMMA_BF16(A, B, C, RA) \
  __builtin_amdgcn_wmma_f32_16x16x32_bf16(false, (A), false, (B), (short)0, (C), (RA), false)

struct Frags { v16bf a, b0, b1, b2, b3; };

#define WMMA_GROUP(F)                              \
  do {                                             \
    acc0 = WMMA_BF16((F).a, (F).b0, acc0, true);   \
    acc1 = WMMA_BF16((F).a, (F).b1, acc1, true);   \
    acc2 = WMMA_BF16((F).a, (F).b2, acc2, true);   \
    acc3 = WMMA_BF16((F).a, (F).b3, acc3, false);  \
  } while (0)

// ---------------- prep kernels ----------------

__global__ void bn_params_kernel(const float* __restrict__ bn,
                                 float* __restrict__ scale,
                                 float* __restrict__ bias, int M) {
  int m = blockIdx.x * blockDim.x + threadIdx.x;
  if (m >= M) return;
  float g = bn[0 * M + m], b = bn[1 * M + m], mu = bn[2 * M + m], v = bn[3 * M + m];
  float s = g * rsqrtf(v + EPSV);
  scale[m] = s;
  bias[m]  = b - mu * s;
}

// NCHW f32 -> NHWC bf16 (thread id = output index, channel fastest)
__global__ void nchw_to_nhwc_bf16(const float* __restrict__ in,
                                  __bf16* __restrict__ out,
                                  int C, int H, int W, int total) {
  int i = blockIdx.x * blockDim.x + threadIdx.x;
  if (i >= total) return;
  int c = i % C; int t = i / C;
  int x = t % W; t /= W;
  int y = t % H; int b = t / H;
  out[i] = (__bf16)in[((b * C + c) * H + y) * W + x];
}

// w[M][C][3][3] f32 -> wp[M][9*C] bf16 with K order k = (dy*3+dx)*C + c
__global__ void fold_w3x3(const float* __restrict__ w, __bf16* __restrict__ wp,
                          int M, int C, int total) {
  int i = blockIdx.x * blockDim.x + threadIdx.x;
  if (i >= total) return;
  int c = i % C; int t = i / C;
  int r = t % 9; int m = t / 9;
  int dy = r / 3, dx = r % 3;
  wp[i] = (__bf16)w[((m * C + c) * 3 + dy) * 3 + dx];
}

// w[M][K] f32 -> wp[Mpad][K] bf16, zero-padded rows for m >= M
__global__ void fold_w1x1(const float* __restrict__ w, __bf16* __restrict__ wp,
                          int M, int Mpad, int K) {
  int i = blockIdx.x * blockDim.x + threadIdx.x;
  if (i >= Mpad * K) return;
  int k = i % K, m = i / K;
  wp[i] = (__bf16)((m < M) ? w[m * K + k] : 0.f);
}

// ---------------- WMMA implicit-GEMM: 3x3 conv + BN + ReLU ----------------
// act: NHWC bf16 [B][H][W][256]; wgt: [256][9*256] tap-major K;
// out: NHWC bf16 [B][OH][OW][256]. One 16x64 strip per wave, 8 waves/block.
__global__ void __launch_bounds__(256) conv3x3_gemm_wmma(
    const __bf16* __restrict__ act, const __bf16* __restrict__ wgt,
    const float* __restrict__ scale, const float* __restrict__ bias,
    __bf16* __restrict__ out, int H, int W, int nQuads) {
  const int OH = H - 2, OW = W - 2;
  const int lane = threadIdx.x & 31;
  const int wave = threadIdx.x >> 5;
  int t = blockIdx.x * 8 + wave;
  if (t >= 16 * nQuads) return;            // wave-uniform
  const int mTile = t & 15;
  const int nQuad = t >> 4;
  const int half = lane >> 4;
  const int l15  = lane & 15;

  const __bf16* wrow = wgt + (size_t)(mTile * 16 + l15) * 2304;

  // Per-column pixel base addresses (dy=dx=0), one per 16-wide sub-tile.
  size_t pixbase[4];
#pragma unroll
  for (int j = 0; j < 4; ++j) {
    int n = nQuad * 64 + j * 16 + l15;
    int b = n / (OH * OW);
    int rm = n - b * (OH * OW);
    int oy = rm / OW, ox = rm - oy * OW;
    pixbase[j] = ((size_t)(b * H + oy) * W + ox) << 8;
  }

  auto load_frags = [&](int k0) {
    int r  = k0 >> 8;                      // tap 0..8 (256 channels/tap)
    int c0 = k0 & 255;
    int dy = r / 3, dx = r - dy * 3;
    size_t doff = ((size_t)(dy * W + dx) << 8) + c0 + half * 8;
    Frags f;
    f.a  = frag_cat(wrow + k0 + half * 8);
    f.b0 = frag_cat(act + pixbase[0] + doff);
    f.b1 = frag_cat(act + pixbase[1] + doff);
    f.b2 = frag_cat(act + pixbase[2] + doff);
    f.b3 = frag_cat(act + pixbase[3] + doff);
    return f;
  };

  v8f acc0 = {}, acc1 = {}, acc2 = {}, acc3 = {};
  Frags fa = load_frags(0);
  int k0 = 0;
  for (; k0 < 2304 - 64; k0 += 64) {       // 35 pairs; K-steps 0..2240
    __builtin_prefetch(wrow + k0 + 64, 0, 1);   // global_prefetch_b8
    Frags fb = load_frags(k0 + 32);
    WMMA_GROUP(fa);
    fa = load_frags(k0 + 64);              // loads define fa directly (no movs)
    WMMA_GROUP(fb);
  }
  {
    Frags fb = load_frags(2304 - 32);
    WMMA_GROUP(fa);
    WMMA_GROUP(fb);
  }

  // Epilogue: lane L, vgpr v -> channel m = 8*half + v, pixel col = l15.
  const int cb = mTile * 16 + half * 8;
  v8f sc = *(const v8f*)(scale + cb);
  v8f bi = *(const v8f*)(bias + cb);
  v8f accs[4] = {acc0, acc1, acc2, acc3};
#pragma unroll
  for (int j = 0; j < 4; ++j) {
    v8bf o;
#pragma unroll
    for (int v = 0; v < 8; ++v) {
      float x = accs[j][v] * sc[v] + bi[v];
      x = fmaxf(x, 0.f);
      o[v] = (__bf16)x;
    }
    int n = nQuad * 64 + j * 16 + l15;
    *(v8bf*)(out + (size_t)n * 256 + cb) = o;
  }
}

// ---------------- depthwise multi-scale xcorr (VALU, ~0.75 GF) ----------------
// s: [64][29][29][256] bf16, kf: [64][5][5][256] bf16
// feat: [64][25][25][768] bf16; taps (5-2i)x(5-2i) offset by i.
__global__ void dwxcorr_kernel(const __bf16* __restrict__ s,
                               const __bf16* __restrict__ kf,
                               __bf16* __restrict__ feat) {
  int idx = blockIdx.x * blockDim.x + threadIdx.x;
  if (idx >= BSZ * 25 * 25 * 3 * 256) return;
  int c = idx & 255; int t = idx >> 8;
  int i = t % 3;  t /= 3;
  int ox = t % 25; t /= 25;
  int oy = t % 25; int b = t / 25;
  int taps = 5 - 2 * i;
  float acc = 0.f;
  for (int ky = 0; ky < taps; ++ky)
    for (int kx = 0; kx < taps; ++kx) {
      float sv = (float)s[((size_t)((b * 29 + oy + i + ky) * 29) + (ox + i + kx)) * 256 + c];
      float kv = (float)kf[((size_t)((b * 5 + i + ky) * 5) + (i + kx)) * 256 + c];
      acc += sv * kv;
    }
  feat[idx] = (__bf16)acc;  // idx == ((b*25+oy)*25+ox)*768 + i*256 + c
}

// ---------------- WMMA GEMM: 1x1 conv + affine (+ReLU), bf16 out ----------------
__global__ void __launch_bounds__(256) gemm1x1_wmma(
    const __bf16* __restrict__ act,   // [N][K]
    const __bf16* __restrict__ wgt,   // [mTiles*16][K]
    const float* __restrict__ scale, const float* __restrict__ bias,
    __bf16* __restrict__ out,         // [N][Cout]
    int K, int mTiles, int nQuads, int Cout, int relu) {
  int lane = threadIdx.x & 31, wave = threadIdx.x >> 5;
  int t = blockIdx.x * 8 + wave;
  if (t >= mTiles * nQuads) return;
  int mTile = t % mTiles, nQuad = t / mTiles;
  int half = lane >> 4, l15 = lane & 15;
  const __bf16* wrow = wgt + (size_t)(mTile * 16 + l15) * K;
  const __bf16* ar0 = act + (size_t)(nQuad * 64 +  0 + l15) * K;
  const __bf16* ar1 = act + (size_t)(nQuad * 64 + 16 + l15) * K;
  const __bf16* ar2 = act + (size_t)(nQuad * 64 + 32 + l15) * K;
  const __bf16* ar3 = act + (size_t)(nQuad * 64 + 48 + l15) * K;

  auto load_frags = [&](int k0) {
    int ko = k0 + half * 8;
    Frags f;
    f.a  = frag_cat(wrow + ko);
    f.b0 = frag_cat(ar0 + ko);
    f.b1 = frag_cat(ar1 + ko);
    f.b2 = frag_cat(ar2 + ko);
    f.b3 = frag_cat(ar3 + ko);
    return f;
  };

  v8f acc0 = {}, acc1 = {}, acc2 = {}, acc3 = {};
  Frags fa = load_frags(0);
  int k0 = 0;
  for (; k0 < K - 64; k0 += 64) {
    __builtin_prefetch(ar0 + k0 + 128, 0, 1);
    Frags fb = load_frags(k0 + 32);
    WMMA_GROUP(fa);
    fa = load_frags(k0 + 64);
    WMMA_GROUP(fb);
  }
  {
    Frags fb = load_frags(K - 32);
    WMMA_GROUP(fa);
    WMMA_GROUP(fb);
  }

  int cb = mTile * 16 + half * 8;
  v8f sc = *(const v8f*)(scale + cb);
  v8f bi = *(const v8f*)(bias + cb);
  v8f accs[4] = {acc0, acc1, acc2, acc3};
#pragma unroll
  for (int j = 0; j < 4; ++j) {
    v8bf o;
#pragma unroll
    for (int v = 0; v < 8; ++v) {
      float x = accs[j][v] * sc[v] + bi[v];
      if (relu) x = fmaxf(x, 0.f);
      o[v] = (__bf16)x;
    }
    int n = nQuad * 64 + j * 16 + l15;
    *(v8bf*)(out + (size_t)n * Cout + cb) = o;
  }
}

// ---------------- final WMMA GEMM: M=10 (padded 16), f32 NCHW out ----------------
__global__ void __launch_bounds__(256) gemm_final_wmma(
    const __bf16* __restrict__ act,   // [40000][256]
    const __bf16* __restrict__ wgt,   // [16][256] zero-padded
    const float* __restrict__ bh2,    // [10]
    float* __restrict__ out,          // [64][10][25][25] NCHW
    int K, int nQuads) {
  int lane = threadIdx.x & 31, wave = threadIdx.x >> 5;
  int t = blockIdx.x * 8 + wave;
  if (t >= nQuads) return;
  int half = lane >> 4, l15 = lane & 15;
  const __bf16* wrow = wgt + (size_t)l15 * K;
  const __bf16* ar0 = act + (size_t)(t * 64 +  0 + l15) * K;
  const __bf16* ar1 = act + (size_t)(t * 64 + 16 + l15) * K;
  const __bf16* ar2 = act + (size_t)(t * 64 + 32 + l15) * K;
  const __bf16* ar3 = act + (size_t)(t * 64 + 48 + l15) * K;

  auto load_frags = [&](int k0) {
    int ko = k0 + half * 8;
    Frags f;
    f.a  = frag_cat(wrow + ko);
    f.b0 = frag_cat(ar0 + ko);
    f.b1 = frag_cat(ar1 + ko);
    f.b2 = frag_cat(ar2 + ko);
    f.b3 = frag_cat(ar3 + ko);
    return f;
  };

  v8f acc0 = {}, acc1 = {}, acc2 = {}, acc3 = {};
  Frags fa = load_frags(0);
  int k0 = 0;
  for (; k0 < K - 64; k0 += 64) {
    Frags fb = load_frags(k0 + 32);
    WMMA_GROUP(fa);
    fa = load_frags(k0 + 64);
    WMMA_GROUP(fb);
  }
  {
    Frags fb = load_frags(K - 32);
    WMMA_GROUP(fa);
    WMMA_GROUP(fb);
  }

  v8f accs[4] = {acc0, acc1, acc2, acc3};
#pragma unroll
  for (int j = 0; j < 4; ++j) {
    int n = t * 64 + j * 16 + l15;
    int b = n / 625, rem = n % 625;   // 25*25
#pragma unroll
    for (int v = 0; v < 8; ++v) {
      int m = half * 8 + v;
      if (m < COUTN) out[((size_t)b * COUTN + m) * 625 + rem] = accs[j][v] + bh2[m];
    }
  }
}

// ---------------- host-side launch ----------------

static inline int cdiv(int a, int b) { return (a + b - 1) / b; }

extern "C" void kernel_launch(void* const* d_in, const int* in_sizes, int n_in,
                              void* d_out, int out_size, void* d_ws, size_t ws_size,
                              hipStream_t stream) {
  const float* in_kernel = (const float*)d_in[0];   // (64,256,7,7)
  const float* in_search = (const float*)d_in[1];   // (64,256,31,31)
  const float* wk  = (const float*)d_in[2];         // (256,256,3,3)
  const float* bnk = (const float*)d_in[3];         // (4,256)
  const float* ws  = (const float*)d_in[4];         // (256,256,3,3)
  const float* bns = (const float*)d_in[5];         // (4,256)
  const float* wd  = (const float*)d_in[6];         // (256,768,1,1)
  const float* bnd = (const float*)d_in[7];         // (4,256)
  const float* wh1 = (const float*)d_in[8];         // (256,256,1,1)
  const float* bnh = (const float*)d_in[9];         // (4,256)
  const float* wh2 = (const float*)d_in[10];        // (10,256,1,1)
  const float* bh2 = (const float*)d_in[11];        // (10,)
  float* out = (float*)d_out;                       // (64,10,25,25)
  (void)in_sizes; (void)n_in; (void)out_size; (void)ws_size;

  // workspace arena (all offsets 256B aligned)
  uintptr_t p = (uintptr_t)d_ws;
  auto take = [&p](size_t bytes) {
    uintptr_t r = p;
    p += (bytes + 255) & ~(size_t)255;
    return r;
  };
  __bf16* kin   = (__bf16*)take((size_t)BSZ * 7 * 7 * 256 * 2);
  __bf16* sin   = (__bf16*)take((size_t)BSZ * 31 * 31 * 256 * 2);
  __bf16* wkp   = (__bf16*)take((size_t)256 * 2304 * 2);
  __bf16* wsp   = (__bf16*)take((size_t)256 * 2304 * 2);
  __bf16* kfeat = (__bf16*)take((size_t)BSZ * 5 * 5 * 256 * 2);
  __bf16* sfeat = (__bf16*)take((size_t)BSZ * 29 * 29 * 256 * 2);
  __bf16* feat  = (__bf16*)take((size_t)BSZ * 25 * 25 * 768 * 2);
  __bf16* wdp   = (__bf16*)take((size_t)256 * 768 * 2);
  __bf16* f2    = (__bf16*)take((size_t)BSZ * 625 * 256 * 2);
  __bf16* wh1p  = (__bf16*)take((size_t)256 * 256 * 2);
  __bf16* hbuf  = (__bf16*)take((size_t)BSZ * 625 * 256 * 2);
  __bf16* wh2p  = (__bf16*)take((size_t)16 * 256 * 2);
  float* sc_k = (float*)take(256 * 4); float* bi_k = (float*)take(256 * 4);
  float* sc_s = (float*)take(256 * 4); float* bi_s = (float*)take(256 * 4);
  float* sc_d = (float*)take(256 * 4); float* bi_d = (float*)take(256 * 4);
  float* sc_h = (float*)take(256 * 4); float* bi_h = (float*)take(256 * 4);

  // --- BN fold params ---
  bn_params_kernel<<<1, 256, 0, stream>>>(bnk, sc_k, bi_k, 256);
  bn_params_kernel<<<1, 256, 0, stream>>>(bns, sc_s, bi_s, 256);
  bn_params_kernel<<<1, 256, 0, stream>>>(bnd, sc_d, bi_d, 256);
  bn_params_kernel<<<1, 256, 0, stream>>>(bnh, sc_h, bi_h, 256);

  // --- weight reorder/convert ---
  {
    int tot = 256 * 9 * 256;
    fold_w3x3<<<cdiv(tot, 256), 256, 0, stream>>>(wk, wkp, 256, 256, tot);
    fold_w3x3<<<cdiv(tot, 256), 256, 0, stream>>>(ws, wsp, 256, 256, tot);
  }
  fold_w1x1<<<cdiv(256 * 768, 256), 256, 0, stream>>>(wd, wdp, 256, 256, 768);
  fold_w1x1<<<cdiv(256 * 256, 256), 256, 0, stream>>>(wh1, wh1p, 256, 256, 256);
  fold_w1x1<<<cdiv(16 * 256, 256), 256, 0, stream>>>(wh2, wh2p, COUTN, 16, 256);

  // --- activation layout: NCHW f32 -> NHWC bf16 ---
  {
    int tot = BSZ * 7 * 7 * 256;
    nchw_to_nhwc_bf16<<<cdiv(tot, 256), 256, 0, stream>>>(in_kernel, kin, 256, 7, 7, tot);
  }
  {
    int tot = BSZ * 31 * 31 * 256;
    nchw_to_nhwc_bf16<<<cdiv(tot, 256), 256, 0, stream>>>(in_search, sin, 256, 31, 31, tot);
  }

  // --- 3x3 convs via WMMA implicit GEMM (fused BN+ReLU) ---
  {
    int nQuads = (BSZ * 5 * 5) / 64;      // 25
    conv3x3_gemm_wmma<<<cdiv(16 * nQuads, 8), 256, 0, stream>>>(
        kin, wkp, sc_k, bi_k, kfeat, 7, 7, nQuads);
  }
  {
    int nQuads = (BSZ * 29 * 29) / 64;    // 841
    conv3x3_gemm_wmma<<<cdiv(16 * nQuads, 8), 256, 0, stream>>>(
        sin, wsp, sc_s, bi_s, sfeat, 31, 31, nQuads);
  }

  // --- depthwise multi-scale cross-correlation ---
  {
    int tot = BSZ * 25 * 25 * 3 * 256;
    dwxcorr_kernel<<<cdiv(tot, 256), 256, 0, stream>>>(sfeat, kfeat, feat);
  }

  // --- 1x1 convs via WMMA GEMM ---
  {
    int nQuads = (BSZ * 625) / 64;        // 625
    gemm1x1_wmma<<<cdiv(16 * nQuads, 8), 256, 0, stream>>>(
        feat, wdp, sc_d, bi_d, f2, 768, 16, nQuads, 256, /*relu=*/0);
    gemm1x1_wmma<<<cdiv(16 * nQuads, 8), 256, 0, stream>>>(
        f2, wh1p, sc_h, bi_h, hbuf, 256, 16, nQuads, 256, /*relu=*/1);
    gemm_final_wmma<<<cdiv(nQuads, 8), 256, 0, stream>>>(
        hbuf, wh2p, bh2, out, 256, nQuads);
  }
}
